// MultiScaleWickRotation_28802050687123
// MI455X (gfx1250) — matrix operand order repaired and checked by reading, not compiled, not run on hardware
//
#include <hip/hip_runtime.h>
#include <hip/hip_bf16.h>

// MultiScaleWickRotation on MI455X (gfx1250):
// 4D FFT -> diag phase -> 4D IFFT as dense 32-point DFT matmuls on the f16
// WMMA path (V_WMMA_F32_16X16X32_F16, K=32 == DFT length).
// The multi-stage transform chain runs register-to-register (symmetric DFT
// matrix => transposed repack of D is a per-lane shuffle). Twiddle tables
// and per-bin rotators are precomputed once into small global tables
// (L0/L2-resident) instead of per-wave transcendentals. Global I/O is
// 128-bit on contiguous planes; the inherent global transpose is 4-byte
// scatter/gather against the 64 MB complex-f16 workspace in the 192 MB L2.

typedef __attribute__((ext_vector_type(16))) _Float16 v16h;
typedef __attribute__((ext_vector_type(8)))  float    v8f;

#define WAVES 8            // 8 independent waves per 256-thread block
#define PASS_TILES 16384   // 16 * 32 * 32
#define PASS_BLOCKS (PASS_TILES / WAVES)

// workspace layout (bytes)
#define WS_SUM_OFF   0
#define WS_TAB_OFF   128        // 3 x 1024 f16 = 6144 B (c, s, -s)
#define WS_ROT_OFF   8192       // 32^4 float2 = 8 MB
#define WS_DATA_OFF  8396800    // 16*32^4 packed complex-f16 = 64 MB

union PK  { unsigned u; _Float16 h[2]; };
union Q16 { uint4 q; _Float16 h[8]; };

__device__ __forceinline__ v8f wmma_f16(v16h a, v16h b, v8f c) {
  // 8 args: (neg_a, A, neg_b, B, c_mod, C, reuse_a, reuse_b)
  return __builtin_amdgcn_wmma_f32_16x16x32_f16(false, a, false, b, (short)0, c,
                                                false, false);
}

// B operand from a SYMMETRIC 32x32 f16 matrix in global memory, row-major:
// B(k,n) = t[k*32+n] = t[n*32+k] -> lane reads k=kb..kb+7 and kb+16..kb+23 from
// row n: 2x global_load_b128 (table is cache-resident, shared by all waves).
__device__ __forceinline__ v16h loadB_g(const _Float16* __restrict__ t,
                                        int n0) {
  const int lane = threadIdx.x & 31;
  const int n    = n0 + (lane & 15);
  const int kb   = (lane >> 4) << 3;
  Q16 lo, hi;
  lo.q = *(const uint4*)(t + n * 32 + kb);
  hi.q = *(const uint4*)(t + n * 32 + kb + 16);
  v16h b;
#pragma unroll
  for (int e = 0; e < 8; ++e) { b[e] = lo.h[e]; b[e + 8] = hi.h[e]; }
  return b;
}

// Repack D accumulators into the next stage's A operand (same lane!):
// A_next(m0'=16j)[e<8]  = f16(D[mblk0][j].v[e])
// A_next(m0'=16j)[e>=8] = f16(D[mblk1][j].v[e-8])
__device__ __forceinline__ v16h packA(v8f lo, v8f hi) {
  v16h a;
#pragma unroll
  for (int e = 0; e < 8; ++e) {
    a[e]     = (_Float16)lo[e];
    a[e + 8] = (_Float16)hi[e];
  }
  return a;
}

// One DFT stage along the column axis of the current (virtual) orientation:
// D = X * W; the transposed repack flips orientation so the next stage along
// the other axis is again a right-multiply. P multiplies Xi in Yr, Q
// multiplies Xr in Yi (forward: P=s, Q=ns ; inverse: P=ns, Q=s).
template <bool IMZ>
__device__ __forceinline__ void stage_reg(v16h* Ar, v16h* Ai, const v16h* Bc,
                                          const v16h* Bp, const v16h* Bq) {
  v8f Dr[2][2], Di[2][2];
#pragma unroll
  for (int m = 0; m < 2; ++m)
#pragma unroll
    for (int n = 0; n < 2; ++n) {
      v8f zz = {};
      v8f aR = zz, aI = zz;
      if constexpr (!IMZ) {
        aR = wmma_f16(Ai[m], Bp[n], zz);
        aI = wmma_f16(Ai[m], Bc[n], zz);
      }
      Dr[m][n] = wmma_f16(Ar[m], Bc[n], aR);
      Di[m][n] = wmma_f16(Ar[m], Bq[n], aI);
    }
  Ar[0] = packA(Dr[0][0], Dr[1][0]);
  Ar[1] = packA(Dr[0][1], Dr[1][1]);
  Ai[0] = packA(Di[0][0], Di[1][0]);
  Ai[1] = packA(Di[0][1], Di[1][1]);
}

// Final stage: input orientation [z][y-freq]; D[m=z][n=y]. Writes phi[y][z]
// straight from the f32 accumulators (64 B contiguous per lane).
__device__ __forceinline__ void stage_reg_out(const v16h* Ar, const v16h* Ai,
                                              const v16h* Bc, const v16h* Bp,
                                              const v16h* Bq,
                                              float2* __restrict__ out) {
  const int lane = threadIdx.x & 31;
#pragma unroll
  for (int m = 0; m < 2; ++m)
#pragma unroll
    for (int n = 0; n < 2; ++n) {
      v8f zz = {};
      v8f Dr = wmma_f16(Ar[m], Bc[n], wmma_f16(Ai[m], Bp[n], zz));
      v8f Di = wmma_f16(Ai[m], Bc[n], wmma_f16(Ar[m], Bq[n], zz));
      const int y  = n * 16 + (lane & 15);
      const int zb = m * 16 + ((lane >> 4) << 3);
#pragma unroll
      for (int v = 0; v < 8; ++v)
        out[y * 32 + zb + v] = make_float2(Dr[v], Di[v]);
    }
}

// A operand straight from a contiguous f32 [32][32] global plane.
__device__ __forceinline__ v16h loadA_g32(const float* __restrict__ g,
                                          int m0) {
  const int lane = threadIdx.x & 31;
  const int m    = m0 + (lane & 15);
  const int co   = (lane >> 4) << 3;
  const float* row = g + m * 32;
  const float4 a0 = *(const float4*)(row + co);
  const float4 a1 = *(const float4*)(row + co + 4);
  const float4 b0 = *(const float4*)(row + co + 16);
  const float4 b1 = *(const float4*)(row + co + 20);
  v16h a;
  a[0]  = (_Float16)a0.x; a[1]  = (_Float16)a0.y;
  a[2]  = (_Float16)a0.z; a[3]  = (_Float16)a0.w;
  a[4]  = (_Float16)a1.x; a[5]  = (_Float16)a1.y;
  a[6]  = (_Float16)a1.z; a[7]  = (_Float16)a1.w;
  a[8]  = (_Float16)b0.x; a[9]  = (_Float16)b0.y;
  a[10] = (_Float16)b0.z; a[11] = (_Float16)b0.w;
  a[12] = (_Float16)b1.x; a[13] = (_Float16)b1.y;
  a[14] = (_Float16)b1.z; a[15] = (_Float16)b1.w;
  return a;
}

// A operands from a contiguous packed-complex-f16 [32][32] workspace plane.
__device__ __forceinline__ void loadA_ws(const unsigned* __restrict__ g,
                                         int m0, v16h& ar, v16h& ai) {
  const int lane = threadIdx.x & 31;
  const int m    = m0 + (lane & 15);
  const int co   = (lane >> 4) << 3;
  const unsigned* row = g + m * 32;
  const uint4 u0 = *(const uint4*)(row + co);
  const uint4 u1 = *(const uint4*)(row + co + 4);
  const uint4 w0 = *(const uint4*)(row + co + 16);
  const uint4 w1 = *(const uint4*)(row + co + 20);
  const unsigned uu[16] = {u0.x, u0.y, u0.z, u0.w, u1.x, u1.y, u1.z, u1.w,
                           w0.x, w0.y, w0.z, w0.w, w1.x, w1.y, w1.z, w1.w};
#pragma unroll
  for (int e = 0; e < 16; ++e) {
    PK p; p.u = uu[e];
    ar[e] = p.h[0];
    ai[e] = p.h[1];
  }
}

// Writeback of the A-layout register state to a contiguous workspace plane.
__device__ __forceinline__ void storeA_ws(unsigned* __restrict__ g, int m0,
                                          v16h ar, v16h ai) {
  const int lane = threadIdx.x & 31;
  const int m    = m0 + (lane & 15);
  const int co   = (lane >> 4) << 3;
  unsigned* row = g + m * 32;
  unsigned uu[16];
#pragma unroll
  for (int e = 0; e < 16; ++e) {
    PK p; p.h[0] = ar[e]; p.h[1] = ai[e];
    uu[e] = p.u;
  }
  uint4 u;
  u.x = uu[0];  u.y = uu[1];  u.z = uu[2];  u.w = uu[3];
  *(uint4*)(row + co) = u;
  u.x = uu[4];  u.y = uu[5];  u.z = uu[6];  u.w = uu[7];
  *(uint4*)(row + co + 4) = u;
  u.x = uu[8];  u.y = uu[9];  u.z = uu[10]; u.w = uu[11];
  *(uint4*)(row + co + 16) = u;
  u.x = uu[12]; u.y = uu[13]; u.z = uu[14]; u.w = uu[15];
  *(uint4*)(row + co + 20) = u;
}

// ---- init: twiddle tables c, s, -s (each 32x32 f16, symmetric) ------------
__global__ __launch_bounds__(256) void init_tables_kernel(
    _Float16* __restrict__ tabs) {
  const int i = blockIdx.x * 256 + threadIdx.x;      // 1024 entries
  const int k = i >> 5, j = i & 31;
  const float inv = 0.17677669529663687f;            // 1/sqrt(32)
  const float ang = 0.19634954084936207f * (float)((k * j) & 31);  // 2pi/32
  float sv, cv;
  __sincosf(ang, &sv, &cv);
  tabs[i]        = (_Float16)(cv * inv);
  tabs[1024 + i] = (_Float16)(sv * inv);
  tabs[2048 + i] = (_Float16)(-sv * inv);
}

// ---- Pass 1: per (b,t,x): fwd DFT over (y,z); scatter to ws[b][ky][kz][t][x]
__global__ __launch_bounds__(256) void pass1_kernel(
    const float* __restrict__ zin, const _Float16* __restrict__ tabs,
    unsigned* __restrict__ ws) {
  const int wv   = threadIdx.x >> 5;
  const int lane = threadIdx.x & 31;
  const int id   = blockIdx.x * WAVES + wv;          // (b*32+t)*32+x
  const int b = id >> 10, t = (id >> 5) & 31, x = id & 31;
  v16h Bc[2]  = {loadB_g(tabs, 0),        loadB_g(tabs, 16)};
  v16h Bs[2]  = {loadB_g(tabs + 1024, 0), loadB_g(tabs + 1024, 16)};
  v16h Bns[2] = {loadB_g(tabs + 2048, 0), loadB_g(tabs + 2048, 16)};
  const float* g = zin + (size_t)id * 1024;          // contiguous (y,z) plane
  v16h Ar[2], Ai[2] = {};
  Ar[0] = loadA_g32(g, 0);
  Ar[1] = loadA_g32(g, 16);
  stage_reg<true>(Ar, Ai, Bc, Bs, Bns);   // fwd z : [y][z]  -> [kz][y]
  stage_reg<false>(Ar, Ai, Bc, Bs, Bns);  // fwd y : [kz][y] -> [ky][kz]
  const int co = (lane >> 4) << 3;
  const size_t base = (size_t)b * 1048576 + t * 32 + x;
#pragma unroll
  for (int blk = 0; blk < 2; ++blk) {
    const int ky = blk * 16 + (lane & 15);
    unsigned* rowp = ws + base + (size_t)ky * 32768;
#pragma unroll
    for (int e = 0; e < 16; ++e) {
      const int kz = (e < 8) ? (co + e) : (16 + co + e - 8);
      PK p; p.h[0] = Ar[blk][e]; p.h[1] = Ai[blk][e];
      rowp[(size_t)kz * 1024] = p.u;                 // L2-resident scatter
    }
  }
}

// ---- Pass 2: per (b,ky,kz): fwd DFT over (t,x), rotate (table lookup),
//      inv DFT, in place
__global__ __launch_bounds__(256) void pass2_kernel(
    unsigned* __restrict__ ws, const _Float16* __restrict__ tabs,
    const float2* __restrict__ rot) {
  const int wv   = threadIdx.x >> 5;
  const int lane = threadIdx.x & 31;
  const int id   = blockIdx.x * WAVES + wv;          // (b*32+ky)*32+kz
  const int kykz = id & 1023;                        // ky*32+kz
  v16h Bc[2]  = {loadB_g(tabs, 0),        loadB_g(tabs, 16)};
  v16h Bs[2]  = {loadB_g(tabs + 1024, 0), loadB_g(tabs + 1024, 16)};
  v16h Bns[2] = {loadB_g(tabs + 2048, 0), loadB_g(tabs + 2048, 16)};
  unsigned* g = ws + (size_t)id * 1024;              // contiguous (t,x) plane
  v16h Ar[2], Ai[2];
  loadA_ws(g, 0, Ar[0], Ai[0]);
  loadA_ws(g, 16, Ar[1], Ai[1]);
  stage_reg<false>(Ar, Ai, Bc, Bs, Bns);  // fwd x : [t][x]  -> [kx][t]
  stage_reg<false>(Ar, Ai, Bc, Bs, Bns);  // fwd t : [kx][t] -> [kt][kx]
  // rotate by e^{i theta}: precomputed rotator table in [ky][kz][kt][kx]
  // order -> this wave reads one contiguous 8 KB plane in 64-B runs.
  const float2* rplane = rot + (size_t)kykz * 1024;
  const int co = (lane >> 4) << 3;
#pragma unroll
  for (int blk = 0; blk < 2; ++blk) {
    const int kt = blk * 16 + (lane & 15);
    const float2* rrow = rplane + kt * 32;
#pragma unroll
    for (int e = 0; e < 16; ++e) {
      const int kx = (e < 8) ? (co + e) : (16 + co + e - 8);
      const float2 cs = rrow[kx];
      const float rr = (float)Ar[blk][e], qq = (float)Ai[blk][e];
      Ar[blk][e] = (_Float16)(rr * cs.x - qq * cs.y);
      Ai[blk][e] = (_Float16)(rr * cs.y + qq * cs.x);
    }
  }
  stage_reg<false>(Ar, Ai, Bc, Bns, Bs);  // inv kx : [kt][kx] -> [x][kt]
  stage_reg<false>(Ar, Ai, Bc, Bns, Bs);  // inv kt : [x][kt]  -> [t][x]
  storeA_ws(g, 0, Ar[0], Ai[0]);
  storeA_ws(g, 16, Ar[1], Ai[1]);
}

// ---- Pass 3: per (b,t,x): gather, inv DFT over (ky,kz), write phi ----------
__global__ __launch_bounds__(256) void pass3_kernel(
    const unsigned* __restrict__ ws, const _Float16* __restrict__ tabs,
    float2* __restrict__ phi) {
  const int wv   = threadIdx.x >> 5;
  const int lane = threadIdx.x & 31;
  const int id   = blockIdx.x * WAVES + wv;          // (b*32+t)*32+x
  const int b = id >> 10, t = (id >> 5) & 31, x = id & 31;
  v16h Bc[2]  = {loadB_g(tabs, 0),        loadB_g(tabs, 16)};
  v16h Bs[2]  = {loadB_g(tabs + 1024, 0), loadB_g(tabs + 1024, 16)};
  v16h Bns[2] = {loadB_g(tabs + 2048, 0), loadB_g(tabs + 2048, 16)};
  const int co = (lane >> 4) << 3;
  const size_t base = (size_t)b * 1048576 + t * 32 + x;
  v16h Ar[2], Ai[2];
#pragma unroll
  for (int blk = 0; blk < 2; ++blk) {
    const int ky = blk * 16 + (lane & 15);
    const unsigned* rowp = ws + base + (size_t)ky * 32768;
#pragma unroll
    for (int e = 0; e < 16; ++e) {
      const int kz = (e < 8) ? (co + e) : (16 + co + e - 8);
      PK p; p.u = rowp[(size_t)kz * 1024];           // L2-resident gather
      Ar[blk][e] = p.h[0];
      Ai[blk][e] = p.h[1];
    }
  }
  stage_reg<false>(Ar, Ai, Bc, Bns, Bs);  // inv kz : [ky][kz] -> [z][ky]
  // inv ky : [z][ky] -> phi[y][z], straight from f32 accumulators
  stage_reg_out(Ar, Ai, Bc, Bns, Bs, phi + (size_t)id * 1024);
}

// ---- theta, rotators, sum(theta), log_det ---------------------------------
__global__ void zero_kernel(float* s) {
  if (threadIdx.x == 0 && blockIdx.x == 0) *s = 0.0f;
}

__global__ __launch_bounds__(256) void theta_kernel(
    const float* __restrict__ tpf, float* __restrict__ theta_out,
    float2* __restrict__ rot, float* __restrict__ sum) {
  __shared__ float red[256];
  const int i = blockIdx.x * 256 + threadIdx.x;      // 32^4 exactly
  const int kt = i >> 15, kx = (i >> 10) & 31, ky = (i >> 5) & 31, kz = i & 31;
  const float a0 = (kt <= 16) ? (float)kt : (float)(kt - 32);
  const float a1 = (kx <= 16) ? (float)kx : (float)(kx - 32);
  const float a2 = (ky <= 16) ? (float)ky : (float)(ky - 32);
  const float a3 = (kz <= 16) ? (float)kz : (float)(kz - 32);
  const float mag = sqrtf(a0 * a0 + a1 * a1 + a2 * a2 + a3 * a3);
  int idx = (int)mag; if (idx > 32) idx = 32;
  const float th = tpf[idx];
  theta_out[i] = th;                                 // [kt][kx][ky][kz]
  float sv, cv;
  __sincosf(th, &sv, &cv);
  // rotator table permuted to [ky][kz][kt][kx] for pass2 locality
  rot[(((size_t)(ky * 32 + kz)) * 32 + kt) * 32 + kx] = make_float2(cv, sv);
  red[threadIdx.x] = th;
  __syncthreads();
  for (int s2 = 128; s2 > 0; s2 >>= 1) {
    if (threadIdx.x < s2) red[threadIdx.x] += red[threadIdx.x + s2];
    __syncthreads();
  }
  if (threadIdx.x == 0) atomicAdd(sum, red[0]);
}

__global__ void logdet_kernel(const float* __restrict__ s,
                              float* __restrict__ out) {
  const int t = threadIdx.x;
  if (t < 16) { out[2 * t] = 0.0f; out[2 * t + 1] = *s; }
}

extern "C" void kernel_launch(void* const* d_in, const int* in_sizes, int n_in,
                              void* d_out, int out_size, void* d_ws,
                              size_t ws_size, hipStream_t stream) {
  (void)in_sizes; (void)n_in; (void)out_size; (void)ws_size;
  const float* zin = (const float*)d_in[0];
  const float* tpf = (const float*)d_in[1];
  float* out = (float*)d_out;

  float*     sum  = (float*)((char*)d_ws + WS_SUM_OFF);
  _Float16*  tabs = (_Float16*)((char*)d_ws + WS_TAB_OFF);
  float2*    rot  = (float2*)((char*)d_ws + WS_ROT_OFF);
  unsigned*  wsd  = (unsigned*)((char*)d_ws + WS_DATA_OFF);

  const size_t phiF = (size_t)16 * 1048576 * 2;      // phi floats
  float* logdet    = out + phiF;                     // 32 floats
  float* theta_out = out + phiF + 32;                // 32^4 floats

  zero_kernel<<<1, 32, 0, stream>>>(sum);
  init_tables_kernel<<<4, 256, 0, stream>>>(tabs);
  theta_kernel<<<4096, 256, 0, stream>>>(tpf, theta_out, rot, sum);
  logdet_kernel<<<1, 32, 0, stream>>>(sum, logdet);

  pass1_kernel<<<PASS_BLOCKS, 256, 0, stream>>>(zin, tabs, wsd);
  pass2_kernel<<<PASS_BLOCKS, 256, 0, stream>>>(wsd, tabs, rot);
  pass3_kernel<<<PASS_BLOCKS, 256, 0, stream>>>(wsd, tabs, out ? (float2*)out
                                                               : (float2*)out);
}
